// TripletLoss_65712999629574
// MI455X (gfx1250) — matrix operand order repaired and checked by reading, not compiled, not run on hardware
//
#include <hip/hip_runtime.h>
#include <hip/hip_bf16.h>

typedef __attribute__((ext_vector_type(16))) __bf16 v16bf;
typedef __attribute__((ext_vector_type(8)))  float  v8f;

#define WMMA_BF16(A, B, C) \
    __builtin_amdgcn_wmma_f32_16x16x32_bf16(false, (A), false, (B), (short)0, (C), false, false)

// ---------------------------------------------------------------------------
// Kernel 0: fp32 -> (bf16 hi, bf16 lo) split + row squared norms
// ---------------------------------------------------------------------------
__global__ __launch_bounds__(256) void tl_convert(const float* __restrict__ X,
                                                  unsigned short* __restrict__ Xhi,
                                                  unsigned short* __restrict__ Xlo,
                                                  float* __restrict__ sq, int d) {
    const int row = blockIdx.x;
    const float* xr = X + (size_t)row * d;
    unsigned short* hr = Xhi + (size_t)row * d;
    unsigned short* lr = Xlo + (size_t)row * d;
    float acc = 0.0f;
    for (int j = threadIdx.x; j < d; j += 256) {
        float x = xr[j];
        acc += x * x;
        unsigned u = __float_as_uint(x);
        unsigned r = u + 0x7FFFu + ((u >> 16) & 1u);        // RNE to bf16
        unsigned short h = (unsigned short)(r >> 16);
        float hf = __uint_as_float(((unsigned)h) << 16);
        float lof = x - hf;
        unsigned ul = __float_as_uint(lof);
        unsigned rl = ul + 0x7FFFu + ((ul >> 16) & 1u);
        hr[j] = h;
        lr[j] = (unsigned short)(rl >> 16);
    }
    __shared__ float s[256];
    s[threadIdx.x] = acc;
    __syncthreads();
    for (int off = 128; off > 0; off >>= 1) {
        if (threadIdx.x < off) s[threadIdx.x] += s[threadIdx.x + off];
        __syncthreads();
    }
    if (threadIdx.x == 0) sq[row] = s[0];
}

// ---------------------------------------------------------------------------
// Kernel 1: Gram via bf16 split WMMA + fused distance epilogue
// Each wave computes a 32x32 tile (2x2 sub-tiles of 16x16).
// ---------------------------------------------------------------------------
__device__ inline v16bf load_a_frag(const unsigned short* base, int half) {
    // A 16x32 bf16: lane holds row M=lane&15; K at half*8..+7 and 16+half*8..+7
    union { uint4 q[2]; v16bf v; } t;
    t.q[0] = *(const uint4*)(base + half * 8);
    t.q[1] = *(const uint4*)(base + 16 + half * 8);
    return t.v;
}
__device__ inline v16bf load_b_frag(const unsigned short* base, int half) {
    // B 32x16 bf16: lane holds col N=lane&15; K at half*8..+7 and 16+half*8..+7
    union { uint4 q[2]; v16bf v; } t;
    const unsigned short* p = base + half * 8;
    t.q[0] = *(const uint4*)(p);
    t.q[1] = *(const uint4*)(p + 16);
    return t.v;
}

__device__ inline void store_dist_tile(v8f acc, int rowBase, int colBase,
                                       const float* __restrict__ sq,
                                       float* __restrict__ dist, int n, int lane) {
    const int nn = lane & 15, half = lane >> 4;
    const int col = colBase + nn;
    const float sqc = sq[col];
#pragma unroll
    for (int v = 0; v < 8; ++v) {
        int row = rowBase + half * 8 + v;      // C layout: VGPR v -> M = v (+8 for hi half)
        float g = acc[v];
        float dd = sq[row] + sqc - 2.0f * g;
        dd = sqrtf(fmaxf(dd, 1e-12f));
        dist[(size_t)row * n + col] = dd;
    }
}

__global__ __launch_bounds__(256) void tl_gram_dist(const unsigned short* __restrict__ Xhi,
                                                    const unsigned short* __restrict__ Xlo,
                                                    const float* __restrict__ sq,
                                                    float* __restrict__ dist,
                                                    int n, int d, int tiles) {
    const int lane = threadIdx.x & 31;
    const int wave = blockIdx.x * 8 + (threadIdx.x >> 5);
    const int tr = wave / tiles, tc = wave % tiles;
    const int rowBase = tr * 32, colBase = tc * 32;
    const int m = lane & 15, half = lane >> 4;

    const unsigned short* a0h = Xhi + (size_t)(rowBase + m) * d;
    const unsigned short* a0l = Xlo + (size_t)(rowBase + m) * d;
    const unsigned short* a1h = Xhi + (size_t)(rowBase + 16 + m) * d;
    const unsigned short* a1l = Xlo + (size_t)(rowBase + 16 + m) * d;
    const unsigned short* b0h = Xhi + (size_t)(colBase + m) * d;
    const unsigned short* b0l = Xlo + (size_t)(colBase + m) * d;
    const unsigned short* b1h = Xhi + (size_t)(colBase + 16 + m) * d;
    const unsigned short* b1l = Xlo + (size_t)(colBase + 16 + m) * d;

    v8f acc00 = {}, acc01 = {}, acc10 = {}, acc11 = {};

    for (int k = 0; k < d; k += 32) {
        v16bf Ah0 = load_a_frag(a0h + k, half);
        v16bf Al0 = load_a_frag(a0l + k, half);
        v16bf Ah1 = load_a_frag(a1h + k, half);
        v16bf Al1 = load_a_frag(a1l + k, half);
        v16bf Bh0 = load_b_frag(b0h + k, half);
        v16bf Bl0 = load_b_frag(b0l + k, half);
        v16bf Bh1 = load_b_frag(b1h + k, half);
        v16bf Bl1 = load_b_frag(b1l + k, half);

        // G ~= hi*hi + hi*lo + lo*hi  (fp32 accumulate)
        acc00 = WMMA_BF16(Ah0, Bh0, acc00);
        acc01 = WMMA_BF16(Ah0, Bh1, acc01);
        acc10 = WMMA_BF16(Ah1, Bh0, acc10);
        acc11 = WMMA_BF16(Ah1, Bh1, acc11);
        acc00 = WMMA_BF16(Ah0, Bl0, acc00);
        acc01 = WMMA_BF16(Ah0, Bl1, acc01);
        acc10 = WMMA_BF16(Ah1, Bl0, acc10);
        acc11 = WMMA_BF16(Ah1, Bl1, acc11);
        acc00 = WMMA_BF16(Al0, Bh0, acc00);
        acc01 = WMMA_BF16(Al0, Bh1, acc01);
        acc10 = WMMA_BF16(Al1, Bh0, acc10);
        acc11 = WMMA_BF16(Al1, Bh1, acc11);
    }

    store_dist_tile(acc00, rowBase,      colBase,      sq, dist, n, lane);
    store_dist_tile(acc01, rowBase,      colBase + 16, sq, dist, n, lane);
    store_dist_tile(acc10, rowBase + 16, colBase,      sq, dist, n, lane);
    store_dist_tile(acc11, rowBase + 16, colBase + 16, sq, dist, n, lane);
}

// ---------------------------------------------------------------------------
// Kernel 2: per-row hardest positive (max, last idx) / hardest negative (min, last idx)
// ---------------------------------------------------------------------------
__global__ __launch_bounds__(256) void tl_reduce1(const float* __restrict__ dist,
                                                  const int* __restrict__ lbl,
                                                  float* __restrict__ ap, float* __restrict__ an,
                                                  int* __restrict__ p, int* __restrict__ q, int n) {
    const int i = blockIdx.x, t = threadIdx.x;
    const int li = lbl[i];
    const float* row = dist + (size_t)i * n;
    float apv = -INFINITY; int api = -1;
    float anv =  INFINITY; int ani = -1;
    for (int j = t; j < n; j += 256) {
        float dv = row[j];
        if (lbl[j] == li) { if (dv >= apv) { apv = dv; api = j; } }
        else              { if (dv <= anv) { anv = dv; ani = j; } }
    }
    __shared__ float sap[256]; __shared__ int iap[256];
    __shared__ float san[256]; __shared__ int ian[256];
    sap[t] = apv; iap[t] = api; san[t] = anv; ian[t] = ani;
    __syncthreads();
    for (int off = 128; off > 0; off >>= 1) {
        if (t < off) {
            if (sap[t + off] > sap[t] || (sap[t + off] == sap[t] && iap[t + off] > iap[t])) {
                sap[t] = sap[t + off]; iap[t] = iap[t + off];
            }
            if (san[t + off] < san[t] || (san[t + off] == san[t] && ian[t + off] > ian[t])) {
                san[t] = san[t + off]; ian[t] = ian[t + off];
            }
        }
        __syncthreads();
    }
    if (t == 0) {
        ap[i] = sap[0];
        an[i] = san[0];
        p[i] = (iap[0] < 0) ? i : iap[0];
        q[i] = (ian[0] < 0) ? i : ian[0];
    }
}

// ---------------------------------------------------------------------------
// Kernel 3: third-class min over dist[q[i],:] + per-row loss
// ---------------------------------------------------------------------------
__global__ __launch_bounds__(256) void tl_reduce2(const float* __restrict__ dist,
                                                  const int* __restrict__ lbl,
                                                  const float* __restrict__ ap,
                                                  const float* __restrict__ an,
                                                  const int* __restrict__ p,
                                                  const int* __restrict__ q,
                                                  float* __restrict__ lossi, int n) {
    const int i = blockIdx.x, t = threadIdx.x;
    const int li = lbl[i];
    const int qi = q[i];
    const int lq = lbl[qi];
    const float* rowq = dist + (size_t)qi * n;
    float mv = INFINITY; int mi = -1;
    for (int j = t; j < n; j += 256) {
        int lj = lbl[j];
        if (lj != li && lj != lq) {
            float dv = rowq[j];
            if (dv <= mv) { mv = dv; mi = j; }
        }
    }
    __shared__ float sm[256]; __shared__ int im[256];
    sm[t] = mv; im[t] = mi;
    __syncthreads();
    for (int off = 128; off > 0; off >>= 1) {
        if (t < off) {
            if (sm[t + off] < sm[t] || (sm[t + off] == sm[t] && im[t + off] > im[t])) {
                sm[t] = sm[t + off]; im[t] = im[t + off];
            }
        }
        __syncthreads();
    }
    if (t == 0) {
        int r = (im[0] < 0) ? 0 : im[0];
        float an4 = rowq[r];                          // dist(hardest neg, third-class)
        float an3 = dist[(size_t)i * n + r];          // dist(anchor, third-class)
        float an2 = dist[(size_t)p[i] * n + qi];      // dist(hardest pos, hardest neg)
        float l = fmaxf(ap[i] - an[i], 0.0f)          // margin ranking, margin=0
                + fabsf(an[i] - an2)
                + fabsf(an3 - an4);
        lossi[i] = l;
    }
}

// ---------------------------------------------------------------------------
// Kernel 4: deterministic final sum -> d_out[0] = sum/n
// ---------------------------------------------------------------------------
__global__ __launch_bounds__(256) void tl_final(const float* __restrict__ lossi,
                                                float* __restrict__ out, int n) {
    const int t = threadIdx.x;
    float acc = 0.0f;
    for (int j = t; j < n; j += 256) acc += lossi[j];
    __shared__ float s[256];
    s[t] = acc;
    __syncthreads();
    for (int off = 128; off > 0; off >>= 1) {
        if (t < off) s[t] += s[t + off];
        __syncthreads();
    }
    if (t == 0) out[0] = s[0] / (float)n;
}

extern "C" void kernel_launch(void* const* d_in, const int* in_sizes, int n_in,
                              void* d_out, int out_size, void* d_ws, size_t ws_size,
                              hipStream_t stream) {
    const float* X   = (const float*)d_in[0];
    const int*   lbl = (const int*)d_in[1];
    const int n = in_sizes[1];
    const int d = in_sizes[0] / n;

    char* ws = (char*)d_ws;
    const size_t xb = (size_t)n * (size_t)d * sizeof(unsigned short);
    const size_t db = (size_t)n * (size_t)n * sizeof(float);
    unsigned short* Xhi  = (unsigned short*)ws;
    unsigned short* Xlo  = (unsigned short*)(ws + xb);
    float*          dist = (float*)(ws + 2 * xb);
    float*          sq   = (float*)(ws + 2 * xb + db);
    float*          ap   = sq + n;
    float*          an   = ap + n;
    int*            p    = (int*)(an + n);
    int*            q    = p + n;
    float*          li   = (float*)(q + n);

    tl_convert<<<n, 256, 0, stream>>>(X, Xhi, Xlo, sq, d);

    const int tiles = n / 32;                 // 128
    const int waves = tiles * tiles;          // 16384 waves, 8 per block
    tl_gram_dist<<<waves / 8, 256, 0, stream>>>(Xhi, Xlo, sq, dist, n, d, tiles);

    tl_reduce1<<<n, 256, 0, stream>>>(dist, lbl, ap, an, p, q, n);
    tl_reduce2<<<n, 256, 0, stream>>>(dist, lbl, ap, an, p, q, li, n);
    tl_final<<<1, 256, 0, stream>>>(li, (float*)d_out, n);
}
